// Batch_Conv1D_87213605912958
// MI455X (gfx1250) — matrix operand order repaired and checked by reading, not compiled, not run on hardware
//
#include <hip/hip_runtime.h>

// ---------------------------------------------------------------------------
// Batched Conv1D (K=3, valid) as implicit-im2col GEMM with bf16 hi/lo-split
// WMMA (v_wmma_f32_16x16x32_bf16) on gfx1250 (wave32).
// x tile staged via Tensor Data Mover (tensor_load_to_lds + s_wait_tensorcnt);
// TDM OOB-zero semantics handle the edge l-tile for free.
// ---------------------------------------------------------------------------

typedef __attribute__((ext_vector_type(16))) __bf16 v16bf;
typedef __attribute__((ext_vector_type(8)))  float  v8f;
typedef __attribute__((ext_vector_type(4)))  unsigned int v4u;
typedef __attribute__((ext_vector_type(8)))  int v8i;
typedef __attribute__((ext_vector_type(4)))  int v4i;

#define L_IN    1024
#define C_IN    256
#define F_OUT   256
#define KW      3
#define L_OUT   1022            // L_IN - KW + 1
#define KC_TOT  768             // KW * C_IN (GEMM K)
#define NB_TOT  256             // 8 * 32 batch-rows

#define MT      64              // l-tile per block
#define FT      64              // f-tile per block
#define XS_ROWS (MT + KW - 1)   // 66 x-rows needed (sliding-window reuse)
#define XS_STRIDE 264           // C_IN + 8 pad (bf16 elems) -> bank spread
#define WT_STRIDE 776           // KC_TOT + 8 pad (bf16 elems)

#define XS_ELEMS (XS_ROWS * XS_STRIDE)   // 17424
#define WT_ELEMS (FT * WT_STRIDE)        // 49664
#define SMEM_BYTES ((2 * XS_ELEMS + 2 * WT_ELEMS) * 2)  // 268,352 B <= 320KB
#define RAW_LDS_OFF (2 * XS_ELEMS * 2)   // 69,696: raw fp32 x tile overlays W region
#define RAW_FLOATS  (XS_ROWS * C_IN)     // 16,896 floats = 67,584 B (fits W region)

// Round-to-nearest-even fp32 -> bf16 hi/lo split: a ~= hi + lo.
__device__ __forceinline__ void bf16_split(float a, unsigned short& h,
                                           unsigned short& l) {
  unsigned int u  = __float_as_uint(a);
  unsigned int rh = u + 0x7FFFu + ((u >> 16) & 1u);
  unsigned short hb = (unsigned short)(rh >> 16);
  float hf  = __uint_as_float(((unsigned int)hb) << 16);
  float res = a - hf;
  unsigned int u2 = __float_as_uint(res);
  unsigned int rl = u2 + 0x7FFFu + ((u2 >> 16) & 1u);
  h = hb;
  l = (unsigned short)(rl >> 16);
}

// A fragment: lane m+16h holds K[16h .. 16h+15] contiguous (32B).
__device__ __forceinline__ v16bf load_frag_a(const unsigned short* p) {
  union { v16bf v; uint4 q[2]; } u;
  u.q[0] = *(const uint4*)(p);
  u.q[1] = *(const uint4*)(p + 8);
  return u.v;
}

// B fragment: lane n+16h holds K[16h .. 16h+15] contiguous (32B).
__device__ __forceinline__ v16bf load_frag_b(const unsigned short* p) {
  union { v16bf v; uint4 q[2]; } u;
  u.q[0] = *(const uint4*)(p);
  u.q[1] = *(const uint4*)(p + 8);
  return u.v;
}

__global__ __launch_bounds__(256, 1)
void conv1d_wmma_bf16split(const float* __restrict__ x,
                           const float* __restrict__ W,
                           const float* __restrict__ bias,
                           float* __restrict__ y) {
  extern __shared__ unsigned short smem[];
  unsigned short* xs_hi = smem;
  unsigned short* xs_lo = xs_hi + XS_ELEMS;
  unsigned short* wt_hi = xs_lo + XS_ELEMS;   // byte offset RAW_LDS_OFF
  unsigned short* wt_lo = wt_hi + WT_ELEMS;

  const int f0  = blockIdx.x * FT;
  const int l0  = blockIdx.y * MT;
  const int nb  = blockIdx.z;
  const int tid = threadIdx.x;
  const int wv  = tid >> 5;

  const float* xn = x + (size_t)nb * (L_IN * C_IN);

  // CDNA5 prefetch hints (global_prefetch_b8) for the tiles we stage next.
  __builtin_prefetch(xn + (size_t)l0 * C_IN, 0, 0);
  __builtin_prefetch(W + f0, 0, 0);

  // ===== Phase 1: raw fp32 x tile (rows l0..l0+65) -> LDS raw buffer =======
  // Raw buffer overlays the (not-yet-used) W region: zero extra LDS.
#if __has_builtin(__builtin_amdgcn_tensor_load_to_lds)
  if (wv == 0) {
    // D# group 0: count=1 | lds_addr | global_addr(57b) | type=2
    unsigned long long ga =
        (unsigned long long)(const void*)(xn + (size_t)l0 * C_IN);
    unsigned int rows_rem = (unsigned int)(L_IN - l0);  // dim1 OOB -> zeros
    v4u g0;
    g0[0] = 1u;                                   // count=1, user mode, no gather
    g0[1] = (unsigned int)RAW_LDS_OFF;            // lds_addr (bytes)
    g0[2] = (unsigned int)(ga & 0xFFFFFFFFu);     // global_addr[31:0]
    g0[3] = (unsigned int)(((ga >> 32) & 0x01FFFFFFu) | (2u << 30)); // +type=2
    // D# group 1: 2D tile [tile_dim0=256, tile_dim1=66] over tensor
    // [tensor_dim0=256, tensor_dim1=rows_rem], dim0 stride=256, data_size=4B.
    v8i g1;
    g1[0] = (int)(2u << 16);                              // data_size code 2
    g1[1] = (int)(((unsigned)C_IN & 0xFFFFu) << 16);      // tensor_dim0 lo16
    g1[2] = (int)((((unsigned)C_IN >> 16) & 0xFFFFu) |
                  ((rows_rem & 0xFFFFu) << 16));          // dim0 hi | dim1 lo
    g1[3] = (int)(((rows_rem >> 16) & 0xFFFFu) |
                  ((unsigned)C_IN << 16));                // dim1 hi | tile_dim0
    g1[4] = (int)XS_ROWS;                                 // tile_dim1=66
    g1[5] = (int)C_IN;                                    // dim0_stride lo32
    g1[6] = 0;
    g1[7] = 0;
    v4i gz4;
    gz4[0] = 0; gz4[1] = 0; gz4[2] = 0; gz4[3] = 0;       // groups 2/3 unused (2D)
    v8i gz8;
    gz8[0] = 0; gz8[1] = 0; gz8[2] = 0; gz8[3] = 0;
    gz8[4] = 0; gz8[5] = 0; gz8[6] = 0; gz8[7] = 0;
    // 6-arg form (clang-23 / therock-10.0 lane): g0, g1, g2, g3, extra, cpol
    __builtin_amdgcn_tensor_load_to_lds(g0, g1, gz4, gz4, gz8, 0);
    __builtin_amdgcn_s_wait_tensorcnt((unsigned short)0);
  }
#else
  {
    float* xraw = (float*)wt_hi;
    for (int idx = tid; idx < RAW_FLOATS / 4; idx += 256) {
      int r = idx / (C_IN / 4);
      int c4 = (idx % (C_IN / 4)) * 4;
      int gl = l0 + r;
      float4 v = make_float4(0.f, 0.f, 0.f, 0.f);
      if (gl < L_IN) v = *(const float4*)(xn + (size_t)gl * C_IN + c4);
      *(float4*)(xraw + idx * 4) = v;
    }
  }
#endif
  __syncthreads();

  // ===== Phase 2: convert raw fp32 tile -> bf16 hi/lo (padded layout) ======
  {
    const float* xraw = (const float*)wt_hi;
    for (int idx = tid; idx < RAW_FLOATS / 4; idx += 256) {
      int r  = idx / (C_IN / 4);
      int c4 = (idx % (C_IN / 4)) * 4;
      float4 v = *(const float4*)(xraw + idx * 4);   // raw is unpadded
      unsigned short h0, h1, h2, h3, lo0, lo1, lo2, lo3;
      bf16_split(v.x, h0, lo0);
      bf16_split(v.y, h1, lo1);
      bf16_split(v.z, h2, lo2);
      bf16_split(v.w, h3, lo3);
      *(ushort4*)(xs_hi + r * XS_STRIDE + c4) = make_ushort4(h0, h1, h2, h3);
      *(ushort4*)(xs_lo + r * XS_STRIDE + c4) = make_ushort4(lo0, lo1, lo2, lo3);
    }
  }
  __syncthreads();   // raw buffer consumed; W region free to overwrite

  // ===== Phase 3: stage W slice transposed to [f][kc], bf16 hi/lo ==========
  for (int idx = tid; idx < KC_TOT * FT; idx += 256) {
    int kc = idx / FT;
    int f  = idx % FT;                       // f fastest -> coalesced reads
    float w = W[(size_t)kc * F_OUT + f0 + f];
    unsigned short h, l;
    bf16_split(w, h, l);
    wt_hi[f * WT_STRIDE + kc] = h;
    wt_lo[f * WT_STRIDE + kc] = l;
  }
  __syncthreads();

  // ===== Phase 4: per-wave 16x32 output tile, K-loop over 768 in 32s =======
  const int lane  = tid & 31;
  const int msub  = wv & 3;          // 0..3  -> rows 16*msub
  const int npair = wv >> 2;         // 0..1  -> cols 32*npair
  const int lr    = lane & 15;
  const int lh    = lane >> 4;

  const int mrow = msub * 16 + lr;   // A matrix row (per-lane M index)
  const unsigned short* b0h = wt_hi + (npair * 32 + lr) * WT_STRIDE;
  const unsigned short* b0l = wt_lo + (npair * 32 + lr) * WT_STRIDE;
  const unsigned short* b1h = b0h + 16 * WT_STRIDE;
  const unsigned short* b1l = b0l + 16 * WT_STRIDE;

  v8f acc0 = {0.f, 0.f, 0.f, 0.f, 0.f, 0.f, 0.f, 0.f};
  v8f acc1 = {0.f, 0.f, 0.f, 0.f, 0.f, 0.f, 0.f, 0.f};

  for (int kc = 0; kc < KC_TOT; kc += 32) {
    // A address: 32-wide K chunk never crosses a tap boundary (256 % 32 == 0)
    const int tap = kc >> 8;
    const int c0  = kc & 255;
    const unsigned short* pa_h =
        xs_hi + (mrow + tap) * XS_STRIDE + c0 + 16 * lh;
    const unsigned short* pa_l =
        xs_lo + (mrow + tap) * XS_STRIDE + c0 + 16 * lh;

    v16bf ahi = load_frag_a(pa_h);
    v16bf alo = load_frag_a(pa_l);
    v16bf bh0 = load_frag_b(b0h + kc + 16 * lh);
    v16bf bl0 = load_frag_b(b0l + kc + 16 * lh);
    v16bf bh1 = load_frag_b(b1h + kc + 16 * lh);
    v16bf bl1 = load_frag_b(b1l + kc + 16 * lh);

    // (ahi+alo)(bhi+blo) ~= ahi*bhi + ahi*blo + alo*bhi   (fp32 accumulate)
    acc0 = __builtin_amdgcn_wmma_f32_16x16x32_bf16(false, ahi, false, bh0,
                                                   (short)0, acc0, false, false);
    acc0 = __builtin_amdgcn_wmma_f32_16x16x32_bf16(false, ahi, false, bl0,
                                                   (short)0, acc0, false, false);
    acc0 = __builtin_amdgcn_wmma_f32_16x16x32_bf16(false, alo, false, bh0,
                                                   (short)0, acc0, false, false);
    acc1 = __builtin_amdgcn_wmma_f32_16x16x32_bf16(false, ahi, false, bh1,
                                                   (short)0, acc1, false, false);
    acc1 = __builtin_amdgcn_wmma_f32_16x16x32_bf16(false, ahi, false, bl1,
                                                   (short)0, acc1, false, false);
    acc1 = __builtin_amdgcn_wmma_f32_16x16x32_bf16(false, alo, false, bh1,
                                                   (short)0, acc1, false, false);
  }

  // ===== store: D VGPR r holds row (r + 8*lh), col lr of the 16x16 tile ====
  const int fg0 = f0 + npair * 32 + lr;
  const int fg1 = fg0 + 16;
  const float bias0 = bias[fg0];
  const float bias1 = bias[fg1];
  float* yn = y + (size_t)nb * (L_OUT * F_OUT);
#pragma unroll
  for (int r = 0; r < 8; ++r) {
    int lg = l0 + msub * 16 + 8 * lh + r;
    if (lg < L_OUT) {
      yn[(size_t)lg * F_OUT + fg0] = acc0[r] + bias0;
      yn[(size_t)lg * F_OUT + fg1] = acc1[r] + bias1;
    }
  }
}

extern "C" void kernel_launch(void* const* d_in, const int* in_sizes, int n_in,
                              void* d_out, int out_size, void* d_ws,
                              size_t ws_size, hipStream_t stream) {
  const float* x = (const float*)d_in[0];   // [8,32,1024,256]
  const float* W = (const float*)d_in[1];   // [3,256,256]
  const float* b = (const float*)d_in[2];   // [256]
  float* y = (float*)d_out;                 // [8,32,1022,256]

  (void)in_sizes; (void)n_in; (void)out_size; (void)d_ws; (void)ws_size;

  // Allow >64KB dynamic LDS (gfx1250 WGP has 320KB). Deterministic host call.
  (void)hipFuncSetAttribute((const void*)conv1d_wmma_bf16split,
                            hipFuncAttributeMaxDynamicSharedMemorySize,
                            SMEM_BYTES);

  dim3 grid(F_OUT / FT, (L_OUT + MT - 1) / MT, NB_TOT);  // (4, 16, 256)
  dim3 block(256);
  conv1d_wmma_bf16split<<<grid, block, SMEM_BYTES, stream>>>(x, W, b, y);
}
